// ABSAHead_89060441850247
// MI455X (gfx1250) — matrix operand order repaired and checked by reading, not compiled, not run on hardware
//
#include <hip/hip_runtime.h>

typedef unsigned short u16;
typedef __attribute__((ext_vector_type(16))) __bf16 v16bf;
typedef __attribute__((ext_vector_type(8)))  float  v8f;

#define N_TOK 8192
#define DIM   512
#define NBRS  12      // adj columns: 2*r(=4) intra + 2*2 leap = 12
#define MTOT  13      // neighbors + self

#define BM    128     // block M tile
#define BN    64      // block N tile
#define BK    64      // K stage
#define STAGES (DIM / BK)   // 8

union BfTile {
  uint4 u[2];
  v16bf v;
};

__device__ __forceinline__ u16 f2bf(float f) {
  unsigned u = __float_as_uint(f);
  unsigned r = 0x7FFFu + ((u >> 16) & 1u);   // round-to-nearest-even
  return (u16)((u + r) >> 16);
}

// Async DMA one 16-byte chunk per lane: global -> LDS (ASYNCcnt-tracked).
__device__ __forceinline__ void async_b128(u16* lds_dst, const u16* g_src) {
  // Low 32 bits of a flat LDS pointer are the LDS byte offset (aperture in high bits).
  unsigned loff = (unsigned)(unsigned long long)lds_dst;
  asm volatile("global_load_async_to_lds_b128 %0, %1, off"
               :: "v"(loff), "v"(g_src)
               : "memory");
}

__device__ __forceinline__ void wait_async_le(int n) {
  if (n == 0)
    asm volatile("s_wait_asynccnt 0" ::: "memory");
  else
    asm volatile("s_wait_asynccnt 6" ::: "memory");
}

// ---------------------------------------------------------------------------
// fp32 -> bf16 staging, 4 elements per thread
// ---------------------------------------------------------------------------
__global__ void cvt_bf16(const float* __restrict__ s, u16* __restrict__ d, int n4) {
  int i = blockIdx.x * blockDim.x + threadIdx.x;
  if (i < n4) {
    float4 v = ((const float4*)s)[i];
    ushort4 o;
    o.x = f2bf(v.x); o.y = f2bf(v.y); o.z = f2bf(v.z); o.w = f2bf(v.w);
    ((ushort4*)d)[i] = o;
  }
}

// ---------------------------------------------------------------------------
// QKV projection: C[mat] = Xb (8192x512) * Wb[mat]^T (512x512), fp32 accum.
// Block (8 waves) computes 128x64 tile; K staged through LDS with async
// global->LDS DMA, double buffered. Each wave -> 16x64 strip via 4 WMMAs
// per 32-K step.
// ---------------------------------------------------------------------------
__global__ __launch_bounds__(256) void qkv_gemm(const u16* __restrict__ Xb,
                                                const u16* __restrict__ Wb,
                                                float* __restrict__ QKV) {
  __shared__ u16 sA[2][BM * BK];   // 2 x 16 KB, row-major [row][k]
  __shared__ u16 sB[2][BN * BK];   // 2 x  8 KB, row-major [col][k]

  const int t    = (int)threadIdx.x;           // 0..255
  const int w    = t >> 5;                     // wave 0..7
  const int lane = t & 31;
  const int lo   = lane & 15;
  const int hi   = lane >> 4;

  const int bx     = (int)blockIdx.x;          // 0..1535
  const int mat    = bx / 512;                 // 0..2 (Q,K,V)
  const int rem    = bx - mat * 512;
  const int mtile  = rem >> 3;                 // 0..63  (128-row tiles)
  const int nstrip = rem & 7;                  // 0..7   (64-col strips)

  const int rows0 = mtile * BM;
  const int col0  = nstrip * BN;

  const u16* Ag = Xb;
  const u16* Bg = Wb + (size_t)mat * DIM * DIM;

  // ---- async stage issue: A = 1024 16B chunks, B = 512 16B chunks --------
  auto issue_stage = [&](int buf, int kk) {
    #pragma unroll
    for (int i = 0; i < 4; ++i) {              // A tile
      int c   = t + i * 256;
      int row = c >> 3, ch = c & 7;
      async_b128(&sA[buf][row * BK + ch * 8],
                 Ag + (size_t)(rows0 + row) * DIM + kk + ch * 8);
    }
    #pragma unroll
    for (int i = 0; i < 2; ++i) {              // B tile
      int c   = t + i * 256;
      int row = c >> 3, ch = c & 7;
      async_b128(&sB[buf][row * BK + ch * 8],
                 Bg + (size_t)(col0 + row) * DIM + kk + ch * 8);
    }
  };

  v8f acc0 = {}, acc1 = {}, acc2 = {}, acc3 = {};
  const int arow = w * 16 + lo;                // this wave's A row for this lane

  issue_stage(0, 0);                           // prologue: stage 0 in flight

  for (int s = 0; s < STAGES; ++s) {
    const int buf = s & 1;
    if (s + 1 < STAGES) {
      issue_stage(buf ^ 1, (s + 1) * BK);      // prefetch next stage
      wait_async_le(6);                        // stage s's 6 DMAs retired
    } else {
      wait_async_le(0);
    }
    __syncthreads();

    #pragma unroll
    for (int kl = 0; kl < BK; kl += 32) {
      BfTile a, b0, b1, b2, b3;
      // A (16x32): lane lo<16 -> row, K {0..15}; hi half K {16..31}
      const u16* ap = &sA[buf][arow * BK + kl + hi * 16];
      a.u[0] = *(const uint4*)ap;
      a.u[1] = *(const uint4*)(ap + 8);
      // B (32x16): lane lo -> col N=lo, K {0..15}; hi half K {16..31}
      const u16* bp0 = &sB[buf][( 0 + lo) * BK + kl + hi * 16];
      const u16* bp1 = &sB[buf][(16 + lo) * BK + kl + hi * 16];
      const u16* bp2 = &sB[buf][(32 + lo) * BK + kl + hi * 16];
      const u16* bp3 = &sB[buf][(48 + lo) * BK + kl + hi * 16];
      b0.u[0] = *(const uint4*)bp0;  b0.u[1] = *(const uint4*)(bp0 + 8);
      b1.u[0] = *(const uint4*)bp1;  b1.u[1] = *(const uint4*)(bp1 + 8);
      b2.u[0] = *(const uint4*)bp2;  b2.u[1] = *(const uint4*)(bp2 + 8);
      b3.u[0] = *(const uint4*)bp3;  b3.u[1] = *(const uint4*)(bp3 + 8);

      acc0 = __builtin_amdgcn_wmma_f32_16x16x32_bf16(false, a.v, false, b0.v,
                                                     (short)0, acc0, false, false);
      acc1 = __builtin_amdgcn_wmma_f32_16x16x32_bf16(false, a.v, false, b1.v,
                                                     (short)0, acc1, false, false);
      acc2 = __builtin_amdgcn_wmma_f32_16x16x32_bf16(false, a.v, false, b2.v,
                                                     (short)0, acc2, false, false);
      acc3 = __builtin_amdgcn_wmma_f32_16x16x32_bf16(false, a.v, false, b3.v,
                                                     (short)0, acc3, false, false);
    }
    __syncthreads();
  }

  // C/D layout: VGPR r, lanes 0-15 -> M=r, N=lane; lanes 16-31 -> M=r+8, N=lane-16
  float* C = QKV + (size_t)mat * N_TOK * DIM
                 + (size_t)(rows0 + w * 16 + hi * 8) * DIM + col0 + lo;
  #pragma unroll
  for (int r = 0; r < 8; ++r) {
    C[(size_t)r * DIM +  0] = acc0[r];
    C[(size_t)r * DIM + 16] = acc1[r];
    C[(size_t)r * DIM + 32] = acc2[r];
    C[(size_t)r * DIM + 48] = acc3[r];
  }
}

// ---------------------------------------------------------------------------
// Sparse attention: one wave32 per token; 16 fp32 elements per lane.
// ---------------------------------------------------------------------------
__global__ __launch_bounds__(256) void absa_attn(const float* __restrict__ Q,
                                                 const float* __restrict__ Km,
                                                 const float* __restrict__ Vm,
                                                 const int* __restrict__ adj,
                                                 float* __restrict__ out) {
  const int tok  = (int)((blockIdx.x * blockDim.x + threadIdx.x) >> 5);
  const int lane = (int)(threadIdx.x & 31);
  const float scale = 0.04419417382415922f;  // 1/sqrt(512)

  const float4* qp = (const float4*)(Q + (size_t)tok * DIM + lane * 16);
  const float4 q0 = qp[0], q1 = qp[1], q2 = qp[2], q3 = qp[3];

  int nbr[MTOT];
  #pragma unroll
  for (int j = 0; j < NBRS; ++j) nbr[j] = adj[tok * NBRS + j];
  nbr[NBRS] = tok;

  float s[MTOT];
  #pragma unroll
  for (int j = 0; j < MTOT; ++j) {
    const float4* kp = (const float4*)(Km + (size_t)nbr[j] * DIM + lane * 16);
    const float4 k0 = kp[0], k1 = kp[1], k2 = kp[2], k3 = kp[3];
    float p = q0.x * k0.x;
    p = fmaf(q0.y, k0.y, p); p = fmaf(q0.z, k0.z, p); p = fmaf(q0.w, k0.w, p);
    p = fmaf(q1.x, k1.x, p); p = fmaf(q1.y, k1.y, p); p = fmaf(q1.z, k1.z, p);
    p = fmaf(q1.w, k1.w, p); p = fmaf(q2.x, k2.x, p); p = fmaf(q2.y, k2.y, p);
    p = fmaf(q2.z, k2.z, p); p = fmaf(q2.w, k2.w, p); p = fmaf(q3.x, k3.x, p);
    p = fmaf(q3.y, k3.y, p); p = fmaf(q3.z, k3.z, p); p = fmaf(q3.w, k3.w, p);
    // wave32 butterfly reduction
    p += __shfl_xor(p, 16, 32);
    p += __shfl_xor(p,  8, 32);
    p += __shfl_xor(p,  4, 32);
    p += __shfl_xor(p,  2, 32);
    p += __shfl_xor(p,  1, 32);
    s[j] = p * scale;
  }

  float mx = s[0];
  #pragma unroll
  for (int j = 1; j < MTOT; ++j) mx = fmaxf(mx, s[j]);
  float sum = 0.f;
  #pragma unroll
  for (int j = 0; j < MTOT; ++j) { s[j] = __expf(s[j] - mx); sum += s[j]; }
  const float inv = 1.f / sum;

  float4 o0 = {0.f, 0.f, 0.f, 0.f}, o1 = o0, o2 = o0, o3 = o0;
  #pragma unroll
  for (int j = 0; j < MTOT; ++j) {
    const float wgt = s[j] * inv;
    const float4* vp = (const float4*)(Vm + (size_t)nbr[j] * DIM + lane * 16);
    const float4 v0 = vp[0], v1 = vp[1], v2 = vp[2], v3 = vp[3];
    o0.x = fmaf(wgt, v0.x, o0.x); o0.y = fmaf(wgt, v0.y, o0.y);
    o0.z = fmaf(wgt, v0.z, o0.z); o0.w = fmaf(wgt, v0.w, o0.w);
    o1.x = fmaf(wgt, v1.x, o1.x); o1.y = fmaf(wgt, v1.y, o1.y);
    o1.z = fmaf(wgt, v1.z, o1.z); o1.w = fmaf(wgt, v1.w, o1.w);
    o2.x = fmaf(wgt, v2.x, o2.x); o2.y = fmaf(wgt, v2.y, o2.y);
    o2.z = fmaf(wgt, v2.z, o2.z); o2.w = fmaf(wgt, v2.w, o2.w);
    o3.x = fmaf(wgt, v3.x, o3.x); o3.y = fmaf(wgt, v3.y, o3.y);
    o3.z = fmaf(wgt, v3.z, o3.z); o3.w = fmaf(wgt, v3.w, o3.w);
  }

  float4* op = (float4*)(out + (size_t)tok * DIM + lane * 16);
  op[0] = o0; op[1] = o1; op[2] = o2; op[3] = o3;
}

// ---------------------------------------------------------------------------
extern "C" void kernel_launch(void* const* d_in, const int* in_sizes, int n_in,
                              void* d_out, int out_size, void* d_ws, size_t ws_size,
                              hipStream_t stream) {
  const float* X  = (const float*)d_in[0];
  const float* Wq = (const float*)d_in[1];
  const float* Wk = (const float*)d_in[2];
  const float* Wv = (const float*)d_in[3];
  const int*   adj = (const int*)d_in[4];
  float* out = (float*)d_out;

  const size_t nX = (size_t)N_TOK * DIM;        // 4,194,304
  const size_t nW = (size_t)DIM * DIM;          //   262,144

  char* wsp = (char*)d_ws;
  u16*  Xb  = (u16*)wsp;                                    // 8 MB
  u16*  Wb  = (u16*)(wsp + nX * sizeof(u16));               // 1.5 MB (q,k,v)
  float* QKV = (float*)(wsp + nX * sizeof(u16) + 3 * nW * sizeof(u16)); // 48 MB

  // stage bf16 operands (4 elems/thread)
  cvt_bf16<<<(int)((nX / 4 + 255) / 256), 256, 0, stream>>>(X,  Xb, (int)(nX / 4));
  cvt_bf16<<<(int)((nW / 4 + 255) / 256), 256, 0, stream>>>(Wq, Wb + 0 * nW, (int)(nW / 4));
  cvt_bf16<<<(int)((nW / 4 + 255) / 256), 256, 0, stream>>>(Wk, Wb + 1 * nW, (int)(nW / 4));
  cvt_bf16<<<(int)((nW / 4 + 255) / 256), 256, 0, stream>>>(Wv, Wb + 2 * nW, (int)(nW / 4));

  // 3 mats * 64 M-tiles(128) * 8 N-strips(64) = 1536 blocks
  qkv_gemm<<<1536, 256, 0, stream>>>(Xb, Wb, QKV);

  const float* Qm = QKV;
  const float* Km = QKV + (size_t)N_TOK * DIM;
  const float* Vm = QKV + 2 * (size_t)N_TOK * DIM;
  absa_attn<<<N_TOK / 8, 256, 0, stream>>>(Qm, Km, Vm, adj, out);
}
